// JambaMambaDecoderLayer_11845519802908
// MI455X (gfx1250) — matrix operand order repaired and checked by reading, not compiled, not run on hardware
//
#include <hip/hip_runtime.h>
#include <hip/hip_bf16.h>
#include <math.h>
#include <stdint.h>

// ---------------- constants ----------------
#define TOKS   4096      // B*L
#define BATCH  2
#define LSEQ   2048
#define DMODEL 1024
#define DINNER 2048
#define DSTATE 16
#define DTRANK 64
#define NEXP   8
#define DFFN   4096
#define DBCW   96        // DTRANK + 2*DSTATE

// ---------------- types ----------------
typedef __attribute__((ext_vector_type(16))) __bf16        v16bf;
typedef __attribute__((ext_vector_type(8)))  float         v8f;
typedef __attribute__((ext_vector_type(4)))  unsigned int  v4u;

union FragU { v16bf v; v4u q[2]; };

__device__ __forceinline__ unsigned short f32_bf16(float f) {
  union { float f; unsigned int u; } c; c.f = f;
  unsigned int r = c.u + 0x7FFFu + ((c.u >> 16) & 1u);   // round-nearest-even
  return (unsigned short)(r >> 16);
}

// A fragment: 16x32 bf16, row = lane&15, K chunks at kb and kb+16 (kb = (lane>>4)*8)
__device__ __forceinline__ v16bf ldfragA(const unsigned short* __restrict__ A,
                                         int lda, int row, int k0, int lane) {
  int kb = (lane >> 4) * 8;
  const unsigned short* p = A + (size_t)row * lda + k0 + kb;
  FragU f;
  f.q[0] = *(const v4u*)(p);
  f.q[1] = *(const v4u*)(p + 16);
  return f.v;
}
// B fragment from transposed weights Bt[N][K]: col = lane&15, K chunks at kb and kb+16 (kb = (lane>>4)*8)
__device__ __forceinline__ v16bf ldfragB(const unsigned short* __restrict__ Bt,
                                         int ldb, int col, int k0, int lane) {
  const unsigned short* p = Bt + (size_t)col * ldb + k0 + ((lane >> 4) * 8);
  FragU f;
  f.q[0] = *(const v4u*)(p);
  f.q[1] = *(const v4u*)(p + 16);
  return f.v;
}

#define WMMA_BF16(A_, B_, C_) \
  __builtin_amdgcn_wmma_f32_16x16x32_bf16(false, (A_), false, (B_), (short)0, (C_), false, false)

__device__ __forceinline__ float silu_f(float x) { return x / (1.f + __expf(-x)); }

// ---------------- weight cast+transpose:  W[K][N] f32 -> Wt[N][K] bf16 ----------------
__global__ __launch_bounds__(256) void k_transpose_cast(const float* __restrict__ W,
                                                        unsigned short* __restrict__ Wt,
                                                        int K, int N) {
  __shared__ float t[32][33];
  size_t boff = (size_t)blockIdx.z * (size_t)K * (size_t)N;
  W  += boff; Wt += boff;
  int n0 = blockIdx.x * 32, k0 = blockIdx.y * 32;
  int tx = threadIdx.x & 31, ty = threadIdx.x >> 5;   // 32 x 8
  for (int r = ty; r < 32; r += 8) {
    int k = k0 + r, n = n0 + tx;
    t[r][tx] = (k < K && n < N) ? W[(size_t)k * N + n] : 0.f;
  }
  __syncthreads();
  for (int r = ty; r < 32; r += 8) {
    int n = n0 + r, k = k0 + tx;
    if (n < N && k < K) Wt[(size_t)n * K + k] = f32_bf16(t[tx][r]);
  }
}

// ---------------- rmsnorm (+ optional f32 out, always bf16 out) ----------------
__global__ __launch_bounds__(256) void k_rmsnorm(const float* __restrict__ x,
                                                 const float* __restrict__ w,
                                                 float* __restrict__ yf,
                                                 unsigned short* __restrict__ yb,
                                                 int D) {
  __shared__ float red[256];
  int tok = blockIdx.x, tid = threadIdx.x;
  const float* xr = x + (size_t)tok * D;
  float ss = 0.f;
  for (int d = tid; d < D; d += 256) { float v = xr[d]; ss += v * v; }
  red[tid] = ss; __syncthreads();
  for (int s = 128; s > 0; s >>= 1) { if (tid < s) red[tid] += red[tid + s]; __syncthreads(); }
  float scale = rsqrtf(red[0] / (float)D + 1e-5f);
  for (int d = tid; d < D; d += 256) {
    float v = xr[d] * scale * w[d];
    if (yf) yf[(size_t)tok * D + d] = v;
    yb[(size_t)tok * D + d] = f32_bf16(v);
  }
}

// ---------------- generic WMMA GEMM: C[M][N](f32) = A_bf16[M][K] @ Bt_bf16[N][K]^T --------
// Wave tile 32x64 (2Mx4N sub-tiles, 8 accums). Block = 4 waves (2x2) -> 64x128 tile.
// EPI: 0 plain | 1 +bias then softplus | 2 +residual | 3 plain + bf16 copy out
template <int EPI>
__global__ __launch_bounds__(128) void k_gemm(const unsigned short* __restrict__ A, int lda,
                                              const unsigned short* __restrict__ Bt, int ldb,
                                              float* __restrict__ C, int ldc,
                                              int N, int K,
                                              const float* __restrict__ extra0,
                                              unsigned short* __restrict__ obf) {
  int lane = threadIdx.x & 31;
  int wave = threadIdx.x >> 5;
  int wm = wave >> 1, wn = wave & 1;
  int m0 = blockIdx.y * 64 + wm * 32;
  int n0 = blockIdx.x * 128 + wn * 64;
  int rA0 = m0 + (lane & 15), rA1 = rA0 + 16;
  int cc[4];
#pragma unroll
  for (int b = 0; b < 4; b++) cc[b] = min(n0 + b * 16 + (lane & 15), N - 1);
  v8f acc[2][4] = {};
  for (int k0 = 0; k0 < K; k0 += 32) {
    v16bf a0 = ldfragA(A, lda, rA0, k0, lane);
    v16bf a1 = ldfragA(A, lda, rA1, k0, lane);
    v16bf b0 = ldfragB(Bt, ldb, cc[0], k0, lane);
    v16bf b1 = ldfragB(Bt, ldb, cc[1], k0, lane);
    v16bf b2 = ldfragB(Bt, ldb, cc[2], k0, lane);
    v16bf b3 = ldfragB(Bt, ldb, cc[3], k0, lane);
    acc[0][0] = WMMA_BF16(a0, b0, acc[0][0]);
    acc[0][1] = WMMA_BF16(a0, b1, acc[0][1]);
    acc[0][2] = WMMA_BF16(a0, b2, acc[0][2]);
    acc[0][3] = WMMA_BF16(a0, b3, acc[0][3]);
    acc[1][0] = WMMA_BF16(a1, b0, acc[1][0]);
    acc[1][1] = WMMA_BF16(a1, b1, acc[1][1]);
    acc[1][2] = WMMA_BF16(a1, b2, acc[1][2]);
    acc[1][3] = WMMA_BF16(a1, b3, acc[1][3]);
  }
#pragma unroll
  for (int a = 0; a < 2; a++) {
    int rbase = m0 + a * 16 + ((lane >> 4) * 8);
#pragma unroll
    for (int b = 0; b < 4; b++) {
      int col = n0 + b * 16 + (lane & 15);
      if (col >= N) continue;
#pragma unroll
      for (int v = 0; v < 8; v++) {
        int row = rbase + v;
        float val = acc[a][b][v];
        if (EPI == 1) { val += extra0[col]; val = (val > 20.f) ? val : log1pf(__expf(val)); }
        if (EPI == 2) { val += extra0[(size_t)row * ldc + col]; }
        C[(size_t)row * ldc + col] = val;
        if (EPI == 3) obf[(size_t)row * ldc + col] = f32_bf16(val);
      }
    }
  }
}

// ---------------- depthwise conv4 + SiLU ----------------
__global__ __launch_bounds__(256) void k_conv(const float* __restrict__ xz,
                                              const float* __restrict__ cw,
                                              const float* __restrict__ cb,
                                              float* __restrict__ xi,
                                              unsigned short* __restrict__ xib) {
  int id = blockIdx.x * 256 + threadIdx.x;
  if (id >= TOKS * DINNER) return;
  int c = id & (DINNER - 1);
  int tok = id >> 11;
  int l = tok & (LSEQ - 1);
  float s = cb[c];
#pragma unroll
  for (int t = 0; t < 4; t++) {
    int ll = l - 3 + t;
    if (ll >= 0) s += xz[(size_t)(tok - (3 - t)) * (2 * DINNER) + c] * cw[c * 4 + t];
  }
  s = silu_f(s);
  xi[id] = s;
  xib[id] = f32_bf16(s);
}

// ---------------- selective scan (sequential over L, parallel over B x DINNER) --------
__global__ __launch_bounds__(128) void k_scan(const float* __restrict__ delta,
                                              const float* __restrict__ dbc,
                                              const float* __restrict__ xi,
                                              const float* __restrict__ xz,
                                              const float* __restrict__ A_log,
                                              const float* __restrict__ Dskip,
                                              unsigned short* __restrict__ ybf) {
  int c = blockIdx.x * 128 + threadIdx.x;
  int b = blockIdx.y;
  float A[DSTATE], h[DSTATE];
#pragma unroll
  for (int n = 0; n < DSTATE; n++) { A[n] = -__expf(A_log[c * DSTATE + n]); h[n] = 0.f; }
  float Dv = Dskip[c];
  __shared__ float bc[16][32];
  for (int l0 = 0; l0 < LSEQ; l0 += 16) {
    for (int i = threadIdx.x; i < 512; i += 128) {
      int s = i >> 5, j = i & 31;
      bc[s][j] = dbc[(size_t)(b * LSEQ + l0 + s) * DBCW + DTRANK + j];
    }
    __syncthreads();
    for (int s = 0; s < 16; s++) {
      int tok = b * LSEQ + l0 + s;
      float d  = delta[(size_t)tok * DINNER + c];
      float x  = xi[(size_t)tok * DINNER + c];
      float dx = d * x;
      float y = 0.f;
#pragma unroll
      for (int n = 0; n < DSTATE; n++) {
        float dA = __expf(d * A[n]);
        h[n] = dA * h[n] + dx * bc[s][n];
        y += h[n] * bc[s][16 + n];
      }
      float z = xz[(size_t)tok * (2 * DINNER) + DINNER + c];
      float yy = (y + Dv * x) * silu_f(z);
      ybf[(size_t)tok * DINNER + c] = f32_bf16(yy);
    }
    __syncthreads();
  }
}

// ---------------- copy residual into d_out, zero expert counters ----------------
__global__ __launch_bounds__(256) void k_copy_init(const float* __restrict__ src,
                                                   float* __restrict__ dst, int n,
                                                   int* __restrict__ cnt) {
  int i = blockIdx.x * 256 + threadIdx.x;
  if (i < n) dst[i] = src[i];
  if (i < NEXP) cnt[i] = 0;
}

// ---------------- router: logits, softmax, top-2, gather lists ----------------
__global__ __launch_bounds__(64) void k_router(const float* __restrict__ h2,
                                               const float* __restrict__ rw,
                                               int* __restrict__ cnt,
                                               int* __restrict__ perm,
                                               float* __restrict__ pw) {
  __shared__ float part[64][NEXP];
  int tok = blockIdx.x, tid = threadIdx.x;
  float acc[NEXP];
#pragma unroll
  for (int e = 0; e < NEXP; e++) acc[e] = 0.f;
  for (int d = tid; d < DMODEL; d += 64) {
    float x = h2[(size_t)tok * DMODEL + d];
    const float* r = rw + d * NEXP;
#pragma unroll
    for (int e = 0; e < NEXP; e++) acc[e] += x * r[e];
  }
#pragma unroll
  for (int e = 0; e < NEXP; e++) part[tid][e] = acc[e];
  __syncthreads();
  if (tid == 0) {
    float lg[NEXP];
#pragma unroll
    for (int e = 0; e < NEXP; e++) lg[e] = 0.f;
    for (int i = 0; i < 64; i++)
#pragma unroll
      for (int e = 0; e < NEXP; e++) lg[e] += part[i][e];
    float m = lg[0];
#pragma unroll
    for (int e = 1; e < NEXP; e++) m = fmaxf(m, lg[e]);
    float s = 0.f, p[NEXP];
#pragma unroll
    for (int e = 0; e < NEXP; e++) { p[e] = __expf(lg[e] - m); s += p[e]; }
#pragma unroll
    for (int e = 0; e < NEXP; e++) p[e] /= s;
    int i1 = 0;
    for (int e = 1; e < NEXP; e++) if (p[e] > p[i1]) i1 = e;
    int i2 = (i1 == 0) ? 1 : 0;
    for (int e = 0; e < NEXP; e++) if (e != i1 && p[e] > p[i2]) i2 = e;
    int pos1 = atomicAdd(&cnt[i1], 1);
    perm[i1 * TOKS + pos1] = tok; pw[i1 * TOKS + pos1] = p[i1];
    int pos2 = atomicAdd(&cnt[i2], 1);
    perm[i2 * TOKS + pos2] = tok; pw[i2 * TOKS + pos2] = p[i2];
  }
}

// ---------------- MoE fused gate+up GEMM (gathered rows), SiLU-mul -> bf16 act --------
__global__ __launch_bounds__(128) void k_moe_gateup(const unsigned short* __restrict__ h2b,
                                                    const unsigned short* __restrict__ WgT,
                                                    const unsigned short* __restrict__ WuT,
                                                    const int* __restrict__ perm,
                                                    const int* __restrict__ cntp,
                                                    unsigned short* __restrict__ act) {
  int cnt = *cntp;
  if ((int)(blockIdx.y * 64) >= cnt) return;
  int lane = threadIdx.x & 31;
  int wave = threadIdx.x >> 5;
  int wm = wave >> 1, wn = wave & 1;
  int m0 = blockIdx.y * 64 + wm * 32;
  if (m0 >= cnt) return;                      // no barriers below; safe divergent exit
  int n0 = blockIdx.x * 64 + wn * 32;
  int sl0 = m0 + (lane & 15), sl1 = sl0 + 16;
  int t0 = perm[min(sl0, cnt - 1)];
  int t1 = perm[min(sl1, cnt - 1)];
  int c0 = n0 + (lane & 15), c1 = c0 + 16;
  v8f g[2][2] = {}, u[2][2] = {};
  for (int k0 = 0; k0 < DMODEL; k0 += 32) {
    v16bf a0  = ldfragA(h2b, DMODEL, t0, k0, lane);
    v16bf a1  = ldfragA(h2b, DMODEL, t1, k0, lane);
    v16bf bg0 = ldfragB(WgT, DMODEL, c0, k0, lane);
    v16bf bg1 = ldfragB(WgT, DMODEL, c1, k0, lane);
    v16bf bu0 = ldfragB(WuT, DMODEL, c0, k0, lane);
    v16bf bu1 = ldfragB(WuT, DMODEL, c1, k0, lane);
    g[0][0] = WMMA_BF16(a0, bg0, g[0][0]);
    g[0][1] = WMMA_BF16(a0, bg1, g[0][1]);
    g[1][0] = WMMA_BF16(a1, bg0, g[1][0]);
    g[1][1] = WMMA_BF16(a1, bg1, g[1][1]);
    u[0][0] = WMMA_BF16(a0, bu0, u[0][0]);
    u[0][1] = WMMA_BF16(a0, bu1, u[0][1]);
    u[1][0] = WMMA_BF16(a1, bu0, u[1][0]);
    u[1][1] = WMMA_BF16(a1, bu1, u[1][1]);
  }
#pragma unroll
  for (int a = 0; a < 2; a++) {
    int rbase = m0 + a * 16 + ((lane >> 4) * 8);
#pragma unroll
    for (int b = 0; b < 2; b++) {
      int col = n0 + b * 16 + (lane & 15);
#pragma unroll
      for (int v = 0; v < 8; v++) {
        int slot = rbase + v;
        if (slot < cnt) {
          float val = silu_f(g[a][b][v]) * u[a][b][v];
          act[(size_t)slot * DFFN + col] = f32_bf16(val);
        }
      }
    }
  }
}

// ---------------- MoE down GEMM + weighted scatter-accumulate into d_out --------------
// Wave tile 32x64 (8 accums), block tile 64x128.
__global__ __launch_bounds__(128) void k_moe_down(const unsigned short* __restrict__ act,
                                                  const unsigned short* __restrict__ WdT,
                                                  const int* __restrict__ perm,
                                                  const float* __restrict__ pw,
                                                  const int* __restrict__ cntp,
                                                  float* __restrict__ out) {
  int cnt = *cntp;
  if ((int)(blockIdx.y * 64) >= cnt) return;
  int lane = threadIdx.x & 31;
  int wave = threadIdx.x >> 5;
  int wm = wave >> 1, wn = wave & 1;
  int m0 = blockIdx.y * 64 + wm * 32;
  if (m0 >= cnt) return;
  int n0 = blockIdx.x * 128 + wn * 64;
  int r0 = min(m0 + (lane & 15), cnt - 1);
  int r1 = min(m0 + (lane & 15) + 16, cnt - 1);
  int cc[4];
#pragma unroll
  for (int b = 0; b < 4; b++) cc[b] = n0 + b * 16 + (lane & 15);
  v8f acc[2][4] = {};
  for (int k0 = 0; k0 < DFFN; k0 += 32) {
    v16bf a0 = ldfragA(act, DFFN, r0, k0, lane);
    v16bf a1 = ldfragA(act, DFFN, r1, k0, lane);
    v16bf b0 = ldfragB(WdT, DFFN, cc[0], k0, lane);
    v16bf b1 = ldfragB(WdT, DFFN, cc[1], k0, lane);
    v16bf b2 = ldfragB(WdT, DFFN, cc[2], k0, lane);
    v16bf b3 = ldfragB(WdT, DFFN, cc[3], k0, lane);
    acc[0][0] = WMMA_BF16(a0, b0, acc[0][0]);
    acc[0][1] = WMMA_BF16(a0, b1, acc[0][1]);
    acc[0][2] = WMMA_BF16(a0, b2, acc[0][2]);
    acc[0][3] = WMMA_BF16(a0, b3, acc[0][3]);
    acc[1][0] = WMMA_BF16(a1, b0, acc[1][0]);
    acc[1][1] = WMMA_BF16(a1, b1, acc[1][1]);
    acc[1][2] = WMMA_BF16(a1, b2, acc[1][2]);
    acc[1][3] = WMMA_BF16(a1, b3, acc[1][3]);
  }
#pragma unroll
  for (int a = 0; a < 2; a++) {
    int rbase = m0 + a * 16 + ((lane >> 4) * 8);
#pragma unroll
    for (int b = 0; b < 4; b++) {
      int col = n0 + b * 16 + (lane & 15);
#pragma unroll
      for (int v = 0; v < 8; v++) {
        int slot = rbase + v;
        if (slot < cnt) {
          int tok = perm[slot];
          float w = pw[slot];
          out[(size_t)tok * DMODEL + col] += w * acc[a][b][v];
        }
      }
    }
  }
}

// ======================= host launcher =======================
extern "C" void kernel_launch(void* const* d_in, const int* in_sizes, int n_in,
                              void* d_out, int out_size, void* d_ws, size_t ws_size,
                              hipStream_t stream) {
  (void)in_sizes; (void)n_in; (void)out_size; (void)ws_size;
  const float* x_in      = (const float*)d_in[0];   // (2,2048,1024)
  const float* rms1_w    = (const float*)d_in[1];
  const float* rms2_w    = (const float*)d_in[2];
  const float* in_proj_w = (const float*)d_in[3];   // (1024,4096)
  const float* conv_w    = (const float*)d_in[4];   // (2048,1,4)
  const float* conv_b    = (const float*)d_in[5];
  const float* x_proj_w  = (const float*)d_in[6];   // (2048,96)
  const float* dt_proj_w = (const float*)d_in[7];   // (64,2048)
  const float* dt_proj_b = (const float*)d_in[8];
  const float* A_log     = (const float*)d_in[9];   // (2048,16)
  const float* D_skip    = (const float*)d_in[10];
  const float* out_proj_w= (const float*)d_in[11];  // (2048,1024)
  const float* router_w  = (const float*)d_in[12];  // (1024,8)
  const float* Wg        = (const float*)d_in[13];  // (8,1024,4096)
  const float* Wu        = (const float*)d_in[14];
  const float* Wd        = (const float*)d_in[15];  // (8,4096,1024)
  float* out = (float*)d_out;

  // workspace carve-out
  char* base = (char*)d_ws;
  size_t off = 0;
  auto alloc = [&](size_t bytes) -> void* {
    void* p = base + off;
    off = (off + bytes + 255) & ~(size_t)255;
    return p;
  };
  unsigned short* wT_in  = (unsigned short*)alloc((size_t)4096 * 1024 * 2);
  unsigned short* wT_x   = (unsigned short*)alloc((size_t)96   * 2048 * 2);
  unsigned short* wT_dt  = (unsigned short*)alloc((size_t)2048 * 64   * 2);
  unsigned short* wT_out = (unsigned short*)alloc((size_t)1024 * 2048 * 2);
  unsigned short* wT_g   = (unsigned short*)alloc((size_t)NEXP * DFFN * DMODEL * 2);
  unsigned short* wT_u   = (unsigned short*)alloc((size_t)NEXP * DFFN * DMODEL * 2);
  unsigned short* wT_d   = (unsigned short*)alloc((size_t)NEXP * DMODEL * DFFN * 2);
  unsigned short* h1b    = (unsigned short*)alloc((size_t)TOKS * DMODEL * 2);
  float*          xz     = (float*)alloc((size_t)TOKS * 2 * DINNER * 4);
  float*          xi     = (float*)alloc((size_t)TOKS * DINNER * 4);
  unsigned short* xib    = (unsigned short*)alloc((size_t)TOKS * DINNER * 2);
  float*          dbc    = (float*)alloc((size_t)TOKS * DBCW * 4);
  unsigned short* dbcb   = (unsigned short*)alloc((size_t)TOKS * DBCW * 2);
  float*          delta  = (float*)alloc((size_t)TOKS * DINNER * 4);
  unsigned short* ybf    = (unsigned short*)alloc((size_t)TOKS * DINNER * 2);
  float*          h_mid  = (float*)alloc((size_t)TOKS * DMODEL * 4);
  float*          h2f    = (float*)alloc((size_t)TOKS * DMODEL * 4);
  unsigned short* h2b    = (unsigned short*)alloc((size_t)TOKS * DMODEL * 2);
  unsigned short* actb   = (unsigned short*)alloc((size_t)TOKS * DFFN * 2);
  int*            perm   = (int*)alloc((size_t)NEXP * TOKS * 4);
  float*          pw     = (float*)alloc((size_t)NEXP * TOKS * 4);
  int*            cnt    = (int*)alloc(NEXP * 4);

  // 1) weight cast + transpose to bf16 [N][K]
  k_transpose_cast<<<dim3(4096/32, 1024/32, 1), 256, 0, stream>>>(in_proj_w, wT_in, 1024, 4096);
  k_transpose_cast<<<dim3(96/32,   2048/32, 1), 256, 0, stream>>>(x_proj_w,  wT_x,  2048, 96);
  k_transpose_cast<<<dim3(2048/32, 64/32,   1), 256, 0, stream>>>(dt_proj_w, wT_dt, 64,   2048);
  k_transpose_cast<<<dim3(1024/32, 2048/32, 1), 256, 0, stream>>>(out_proj_w,wT_out,2048, 1024);
  k_transpose_cast<<<dim3(4096/32, 1024/32, NEXP), 256, 0, stream>>>(Wg, wT_g, 1024, 4096);
  k_transpose_cast<<<dim3(4096/32, 1024/32, NEXP), 256, 0, stream>>>(Wu, wT_u, 1024, 4096);
  k_transpose_cast<<<dim3(1024/32, 4096/32, NEXP), 256, 0, stream>>>(Wd, wT_d, 4096, 1024);

  // 2) rmsnorm1 -> bf16 activations
  k_rmsnorm<<<TOKS, 256, 0, stream>>>(x_in, rms1_w, nullptr, h1b, DMODEL);

  // 3) in_proj: xz[4096][4096] = h1 @ in_proj_w   (block tile 64x128)
  k_gemm<0><<<dim3(4096/128, TOKS/64), 128, 0, stream>>>(h1b, DMODEL, wT_in, DMODEL,
                                                         xz, 2*DINNER, 2*DINNER, DMODEL,
                                                         nullptr, nullptr);
  // 4) depthwise conv + SiLU
  k_conv<<<(TOKS*DINNER)/256, 256, 0, stream>>>(xz, conv_w, conv_b, xi, xib);

  // 5) x_proj: dbc[4096][96] (f32 + bf16) — one 128-wide N tile, clamped
  k_gemm<3><<<dim3(1, TOKS/64), 128, 0, stream>>>(xib, DINNER, wT_x, DINNER,
                                                  dbc, DBCW, DBCW, DINNER,
                                                  nullptr, dbcb);
  // 6) dt_proj + bias + softplus -> delta[4096][2048]
  k_gemm<1><<<dim3(DINNER/128, TOKS/64), 128, 0, stream>>>(dbcb, DBCW, wT_dt, DTRANK,
                                                           delta, DINNER, DINNER, DTRANK,
                                                           dt_proj_b, nullptr);
  // 7) selective scan (+D skip, * silu(z)) -> y bf16
  k_scan<<<dim3(DINNER/128, BATCH), 128, 0, stream>>>(delta, dbc, xi, xz, A_log, D_skip, ybf);

  // 8) out_proj + residual -> h_mid
  k_gemm<2><<<dim3(DMODEL/128, TOKS/64), 128, 0, stream>>>(ybf, DINNER, wT_out, DINNER,
                                                           h_mid, DMODEL, DMODEL, DINNER,
                                                           x_in, nullptr);
  // 9) seed d_out with residual, zero expert counters
  k_copy_init<<<(TOKS*DMODEL)/256, 256, 0, stream>>>(h_mid, out, TOKS*DMODEL, cnt);

  // 10) rmsnorm2 -> h2 (f32 for router, bf16 for expert GEMMs)
  k_rmsnorm<<<TOKS, 256, 0, stream>>>(h_mid, rms2_w, h2f, h2b, DMODEL);

  // 11) router softmax top-2 -> per-expert gather lists
  k_router<<<TOKS, 64, 0, stream>>>(h2f, router_w, cnt, perm, pw);

  // 12) experts, sequential (deterministic per-token accumulation order)
  for (int e = 0; e < NEXP; e++) {
    const unsigned short* wg = wT_g + (size_t)e * DFFN * DMODEL;
    const unsigned short* wu = wT_u + (size_t)e * DFFN * DMODEL;
    const unsigned short* wd = wT_d + (size_t)e * DMODEL * DFFN;
    k_moe_gateup<<<dim3(DFFN/64, TOKS/64), 128, 0, stream>>>(h2b, wg, wu,
                                                             perm + e*TOKS, cnt + e, actb);
    k_moe_down<<<dim3(DMODEL/128, TOKS/64), 128, 0, stream>>>(actb, wd,
                                                              perm + e*TOKS, pw + e*TOKS,
                                                              cnt + e, out);
  }
}
